// MahaClassifierOriginal_53584011985261
// MI455X (gfx1250) — compile-verified, matrix operand order checked by reading, not run on hardware
//
#include <hip/hip_runtime.h>
#include <hip/hip_bf16.h>
#include <stdint.h>

#define NCLS 64
#define SHOTS 16
#define D 512
#define NCTX (NCLS * SHOTS)   // 1024
#define NTGT 8192
#define LDD 520               // padded bf16 row stride for the diff tile (bank-conflict avoidance)

// k_logits tiling
#define BT 128                // targets per block
#define MT 8                  // 16-row M-tiles per block (all waves sweep all of them)
#define NWAVES 8              // 256 threads
#define NTPW 4                // N-tiles owned per wave = 32 / NWAVES

typedef __bf16 v16bf __attribute__((ext_vector_type(16)));
typedef float  v8f   __attribute__((ext_vector_type(8)));

static __device__ __forceinline__ unsigned short f2bf(float f) {
    union { float f; uint32_t u; } x; x.f = f;
    uint32_t u = x.u;
    uint32_t r = (u + 0x7FFFu + ((u >> 16) & 1u)) >> 16;   // RNE
    return (unsigned short)r;
}
static __device__ __forceinline__ float b2f(unsigned short h) {
    union { uint32_t u; float f; } x; x.u = ((uint32_t)h) << 16;
    return x.f;
}

// ---------------------------------------------------------------------------
// K1: per-class counts, row lists, means (also written to d_out tail)
// ---------------------------------------------------------------------------
__global__ void k_class_stats(const float* __restrict__ ctx, const int* __restrict__ lab,
                              float* __restrict__ means, float* __restrict__ counts,
                              int* __restrict__ classrows, float* __restrict__ outMeans) {
    int c = blockIdx.x;
    __shared__ int labLds[NCTX];
    __shared__ int rowsLds[SHOTS];
    __shared__ int cntS;
    for (int i = threadIdx.x; i < NCTX; i += blockDim.x) labLds[i] = lab[i];
    __syncthreads();
    if (threadIdx.x == 0) {
        for (int k = 0; k < SHOTS; ++k) rowsLds[k] = 0;
        int s = 0;
        for (int r = 0; r < NCTX; ++r)
            if (labLds[r] == c && s < SHOTS) rowsLds[s++] = r;
        cntS = s;
        counts[c] = (float)s;
        for (int k = 0; k < SHOTS; ++k) classrows[c * SHOTS + k] = rowsLds[k];
    }
    __syncthreads();
    int n = cntS;
    float inv = 1.0f / (float)n;
    for (int col = threadIdx.x; col < D; col += blockDim.x) {
        float s = 0.f;
        for (int k = 0; k < n; ++k) s += ctx[(size_t)rowsLds[k] * D + col];
        float m = s * inv;
        means[(size_t)c * D + col]    = m;
        outMeans[(size_t)c * D + col] = m;
    }
}

// ---------------------------------------------------------------------------
// K2: global column mean of context features
// ---------------------------------------------------------------------------
__global__ void k_gmean(const float* __restrict__ ctx, float* __restrict__ gmean) {
    int col = blockIdx.x * blockDim.x + threadIdx.x;
    if (col >= D) return;
    float s = 0.f;
    for (int r = 0; r < NCTX; ++r) s += ctx[(size_t)r * D + col];
    gmean[col] = s * (1.0f / (float)NCTX);
}

// ---------------------------------------------------------------------------
// K3: task covariance (LDS-tiled outer-product accumulation)
// ---------------------------------------------------------------------------
__global__ void k_taskcov(const float* __restrict__ ctx, const float* __restrict__ gmean,
                          float* __restrict__ taskcov) {
    __shared__ float xi[16][17], xj[16][17];
    int tx = threadIdx.x, ty = threadIdx.y;
    int i0 = blockIdx.y * 16, j0 = blockIdx.x * 16;
    float gmi = gmean[i0 + tx], gmj = gmean[j0 + tx];
    float s = 0.f;
    for (int r0 = 0; r0 < NCTX; r0 += 16) {
        xi[ty][tx] = ctx[(size_t)(r0 + ty) * D + i0 + tx] - gmi;
        xj[ty][tx] = ctx[(size_t)(r0 + ty) * D + j0 + tx] - gmj;
        __syncthreads();
        #pragma unroll
        for (int rr = 0; rr < 16; ++rr) s += xi[rr][ty] * xj[rr][tx];
        __syncthreads();
    }
    taskcov[(size_t)(i0 + ty) * D + (j0 + tx)] = s * (1.0f / (float)(NCTX - 1));
}

// ---------------------------------------------------------------------------
// K4: build regularized per-class covariance:
//     reg_c = lam*class_cov + (1-lam)*task_cov + I
// ---------------------------------------------------------------------------
__global__ void k_buildreg(const float* __restrict__ ctx, const int* __restrict__ classrows,
                           const float* __restrict__ counts, const float* __restrict__ means,
                           const float* __restrict__ taskcov, float* __restrict__ regAll) {
    int c = blockIdx.z;
    __shared__ float xi[SHOTS][17], xj[SHOTS][17];
    __shared__ int rows[SHOTS];
    int tx = threadIdx.x, ty = threadIdx.y;
    if (ty == 0 && tx < SHOTS) rows[tx] = classrows[c * SHOTS + tx];
    __syncthreads();
    int i0 = blockIdx.y * 16, j0 = blockIdx.x * 16;
    xi[ty][tx] = ctx[(size_t)rows[ty] * D + i0 + tx];
    xj[ty][tx] = ctx[(size_t)rows[ty] * D + j0 + tx];
    __syncthreads();
    float nc = counts[c];
    int n = (int)nc;
    int gi = i0 + ty, gj = j0 + tx;
    float s = 0.f;
    for (int k = 0; k < n; ++k) s += xi[k][ty] * xj[k][tx];
    float mi = means[(size_t)c * D + gi];
    float mj = means[(size_t)c * D + gj];
    float ccov = (s - nc * mi * mj) / (nc - 1.0f);
    float lam  = nc / (nc + 1.0f);
    float v = lam * ccov + (1.0f - lam) * taskcov[(size_t)gi * D + gj]
            + ((gi == gj) ? 1.0f : 0.0f);
    regAll[((size_t)c * D + gi) * D + gj] = v;
}

// ---------------------------------------------------------------------------
// K5: in-place Gauss-Jordan inversion (f32), one workgroup per class.
// SPD + I => diagonally positive, no pivoting needed. Column-parallel so all
// matrix reads/writes are coalesced; pivot row/column staged in LDS.
// ---------------------------------------------------------------------------
__global__ __launch_bounds__(D) void k_gj(float* __restrict__ regAll) {
    int c = blockIdx.x;
    float* A = regAll + (size_t)c * D * D;
    __shared__ float prow[D];
    __shared__ float fcol[D];
    int t = threadIdx.x;   // column owned by this thread
    for (int k = 0; k < D; ++k) {
        __syncthreads();                       // updates of iter k-1 visible
        prow[t] = A[(size_t)k * D + t];        // pivot row (unscaled)
        fcol[t] = A[(size_t)t * D + k];        // pivot column (old values)
        __syncthreads();
        float ip = 1.0f / prow[k];
        float pr = (t == k) ? ip : prow[t] * ip;
        for (int i = 0; i < D; ++i) {
            float f = fcol[i];
            float* e = &A[(size_t)i * D + t];
            if (i == k)       *e = pr;
            else if (t == k)  *e = -f * ip;
            else              *e = *e - f * pr;
        }
    }
}

// ---------------------------------------------------------------------------
// K6: convert precision matrices f32 -> bf16
// ---------------------------------------------------------------------------
__global__ void k_cvt(const float* __restrict__ p, unsigned short* __restrict__ pb, size_t n) {
    size_t i = (size_t)blockIdx.x * blockDim.x + threadIdx.x;
    size_t stride = (size_t)gridDim.x * blockDim.x;
    for (; i < n; i += stride) pb[i] = f2bf(p[i]);
}

// ---------------------------------------------------------------------------
// K7: fused Mahalanobis logits via WMMA bf16->f32.
// Block = (class c, 128 targets), 8 waves. The 32 N-tiles are split across
// waves (no redundant B traffic); each wave holds 8 M-tile accumulators so
// every B fragment (row-panel of the symmetric P, loaded from L2) is reused
// by 8 WMMAs. The kk loop is kept rolled (#pragma unroll 1): letting the
// compiler unroll it makes it hoist all mt*kk A-fragments across the ntI
// loop (they are ntI-invariant) and spill 4KB/lane to scratch.
// logits = -rowsum(Y .* Diff) fused per N-tile; cross-wave reduction via
// per-wave LDS slices summed in fixed order (deterministic).
// ---------------------------------------------------------------------------
__global__ __launch_bounds__(256) void k_logits(const float* __restrict__ tgt,
                                                const float* __restrict__ means,
                                                const unsigned short* __restrict__ pbf,
                                                float* __restrict__ out) {
    int c = blockIdx.y;
    int tBase = blockIdx.x * BT;
    __shared__ unsigned short diff[BT * LDD];      // 133,120 B
    __shared__ float meanLds[D];                   // 2 KB
    __shared__ float red[NWAVES * BT];             // 4 KB
    int tid = threadIdx.x;

    for (int i = tid; i < D; i += 256) meanLds[i] = means[(size_t)c * D + i];
    __syncthreads();
    for (int idx = tid; idx < BT * D; idx += 256) {
        int r = idx >> 9, col = idx & (D - 1);
        float v = meanLds[col] - tgt[(size_t)(tBase + r) * D + col];
        diff[r * LDD + col] = f2bf(v);
    }
    __syncthreads();

    int wave = tid >> 5, lane = tid & 31;
    int lrow  = lane & 15;
    int kbase = (lane < 16) ? 0 : 8;
    int mAdd  = (lane < 16) ? 0 : 8;
    const unsigned short* Pc = pbf + (size_t)c * D * D;

    float partial[MT][8];
    #pragma unroll
    for (int mt = 0; mt < MT; ++mt)
        #pragma unroll
        for (int r = 0; r < 8; ++r) partial[mt][r] = 0.f;

    #pragma unroll 1
    for (int ntI = 0; ntI < NTPW; ++ntI) {
        int nt = wave + ntI * NWAVES;              // this wave's N-tile
        const unsigned short* brow = Pc + (size_t)(nt * 16 + lrow) * D;  // symmetry: col n == row n
        if (ntI + 1 < NTPW) {                      // prefetch next N-panel of P into L2/L0
            const unsigned short* nrow = Pc + (size_t)((nt + NWAVES) * 16 + lrow) * D;
            __builtin_prefetch(nrow, 0, 3);
            __builtin_prefetch(nrow + D / 2, 0, 3);
        }
        v8f acc[MT];
        #pragma unroll
        for (int mt = 0; mt < MT; ++mt)
            acc[mt] = (v8f){0.f, 0.f, 0.f, 0.f, 0.f, 0.f, 0.f, 0.f};

        #pragma unroll 1                            // keep rolled: prevents A-frag hoist + spill
        for (int kk = 0; kk < 16; ++kk) {
            int k0 = kk * 32;
            union { v16bf v; uint32_t u[8]; } Bf;
            #pragma unroll
            for (int p = 0; p < 4; ++p) {
                Bf.u[p]     = *(const uint32_t*)&brow[k0 + kbase + 2 * p];
                Bf.u[4 + p] = *(const uint32_t*)&brow[k0 + 16 + kbase + 2 * p];
            }
            #pragma unroll
            for (int mt = 0; mt < MT; ++mt) {
                const unsigned short* arow = &diff[(mt * 16 + lrow) * LDD];
                union { v16bf v; uint32_t u[8]; } Af;
                #pragma unroll
                for (int p = 0; p < 4; ++p) {
                    Af.u[p]     = *(const uint32_t*)&arow[k0 + kbase + 2 * p];
                    Af.u[4 + p] = *(const uint32_t*)&arow[k0 + 16 + kbase + 2 * p];
                }
                acc[mt] = __builtin_amdgcn_wmma_f32_16x16x32_bf16(
                              false, Af.v, false, Bf.v, (short)0, acc[mt], false, false);
            }
        }
        // fuse rowsum(Y .* Diff): lane holds N = nt*16+lrow, M = r + mAdd per mt
        int nCol = nt * 16 + lrow;
        #pragma unroll
        for (int mt = 0; mt < MT; ++mt) {
            #pragma unroll
            for (int r = 0; r < 8; ++r) {
                float dv = b2f(diff[(mt * 16 + r + mAdd) * LDD + nCol]);
                partial[mt][r] += acc[mt][r] * dv;
            }
        }
    }

    // reduce across the 16 lanes sharing the same M (xor of lane bits 0..3)
    #pragma unroll
    for (int mt = 0; mt < MT; ++mt) {
        #pragma unroll
        for (int r = 0; r < 8; ++r) {
            float v = partial[mt][r];
            v += __shfl_xor(v, 1, 32);
            v += __shfl_xor(v, 2, 32);
            v += __shfl_xor(v, 4, 32);
            v += __shfl_xor(v, 8, 32);
            partial[mt][r] = v;
        }
    }
    // each wave deposits its per-target partial sums (over its N subset)
    if (lrow == 0) {
        #pragma unroll
        for (int mt = 0; mt < MT; ++mt)
            #pragma unroll
            for (int r = 0; r < 8; ++r)
                red[wave * BT + mt * 16 + r + mAdd] = partial[mt][r];
    }
    __syncthreads();
    // fixed-order cross-wave sum -> deterministic output
    for (int t = tid; t < BT; t += 256) {
        float s = 0.f;
        #pragma unroll
        for (int w = 0; w < NWAVES; ++w) s += red[w * BT + t];
        out[(size_t)(tBase + t) * NCLS + c] = -s;
    }
}

// ---------------------------------------------------------------------------
extern "C" void kernel_launch(void* const* d_in, const int* in_sizes, int n_in,
                              void* d_out, int out_size, void* d_ws, size_t ws_size,
                              hipStream_t stream) {
    (void)in_sizes; (void)n_in; (void)out_size; (void)ws_size;
    const float* ctx = (const float*)d_in[0];
    const int*   lab = (const int*)d_in[1];
    const float* tgt = (const float*)d_in[2];

    float* logitsOut = (float*)d_out;                       // [8192, 64]
    float* outMeans  = (float*)d_out + (size_t)NTGT * NCLS; // [64, 512]

    char* w = (char*)d_ws;
    float* gmean     = (float*)w;  w += (size_t)D * 4;
    float* counts    = (float*)w;  w += (size_t)NCLS * 4;
    float* means     = (float*)w;  w += (size_t)NCLS * D * 4;
    int*   classrows = (int*)w;    w += (size_t)NCLS * SHOTS * 4;
    float* taskcov   = (float*)w;  w += (size_t)D * D * 4;
    float* regAll    = (float*)w;  w += (size_t)NCLS * D * D * 4;   // 64 MB, becomes P (f32)
    unsigned short* pbf = (unsigned short*)w;                        // 32 MB bf16 copy

    k_class_stats<<<NCLS, 256, 0, stream>>>(ctx, lab, means, counts, classrows, outMeans);
    k_gmean<<<(D + 255) / 256, 256, 0, stream>>>(ctx, gmean);
    k_taskcov<<<dim3(D / 16, D / 16), dim3(16, 16), 0, stream>>>(ctx, gmean, taskcov);
    k_buildreg<<<dim3(D / 16, D / 16, NCLS), dim3(16, 16), 0, stream>>>(
        ctx, classrows, counts, means, taskcov, regAll);
    k_gj<<<NCLS, D, 0, stream>>>(regAll);
    k_cvt<<<4096, 256, 0, stream>>>(regAll, pbf, (size_t)NCLS * D * D);
    k_logits<<<dim3(NTGT / BT, NCLS), 256, 0, stream>>>(tgt, means, pbf, logitsOut);
}
